// PropMLP_22428319220067
// MI455X (gfx1250) — compile-verified, matrix operand order checked by reading, not run on hardware
//
#include <hip/hip_runtime.h>

typedef __attribute__((ext_vector_type(16))) _Float16 v16h;
typedef __attribute__((ext_vector_type(8)))  _Float16 v8h;
typedef __attribute__((ext_vector_type(8)))  float    v8f;

#define NLEV   10
#define TSIZE  65536
#define DIN    20      // NUM_LEVELS * FEATURES
#define KPAD   32      // pad K to the WMMA K=32 shape
#define WIDTH  64      // hidden width
#define BPTS   256     // points per block (= threads per block)

// ds_swizzle_b32 group-of-32 xor pattern: offset = and_mask(0x1F) | xor_mask<<10
#define SWZ_XOR(x, mask) \
    __int_as_float(__builtin_amdgcn_ds_swizzle(__float_as_int(x), (0x1F | ((mask) << 10))))

// np.round(exp(linspace(log 16, log 512, 10)))
__device__ __constant__ float GRID_SIZES[NLEV] =
    {16.f, 24.f, 35.f, 51.f, 75.f, 110.f, 161.f, 237.f, 348.f, 512.f};

__global__ __launch_bounds__(BPTS)
void ngp_fused_kernel(const float* __restrict__ pos,
                      const float* __restrict__ table,
                      const float* __restrict__ W1,
                      const float* __restrict__ b1,
                      const float* __restrict__ W2,
                      const float* __restrict__ b2,
                      float* __restrict__ out,
                      int N)
{
    // A-operand staging: one zero-padded 32 x f16 row per point (16 KB)
    __shared__ __align__(16) _Float16 sA[BPTS * KPAD];
    // LDS-staged weights: W1 (5 KB) + b1|W2 (512 B); loaded coalesced once per block
    __shared__ float sW1[DIN * WIDTH];
    __shared__ float sB1W2[2 * WIDTH];

    const int t     = threadIdx.x;
    const int pbase = blockIdx.x * BPTS;
    const int p     = pbase + t;

    // ------------- Stage weights to LDS (coalesced, once per block) -------------
    #pragma unroll
    for (int i = t; i < DIN * WIDTH; i += BPTS) sW1[i] = W1[i];
    if (t < WIDTH) {
        sB1W2[t]         = b1[t];
        sB1W2[WIDTH + t] = W2[t];
    }

    // ---------------- Phase 1: hash-grid encode -> LDS (f16) ----------------
    unsigned* sA32 = reinterpret_cast<unsigned*>(sA);
    if (p < N) {
        const float fx = pos[3 * p + 0];
        const float fy = pos[3 * p + 1];
        const float fz = pos[3 * p + 2];

        #pragma unroll
        for (int l = 0; l < NLEV; ++l) {
            const float g  = GRID_SIZES[l];
            const float sx = fx * g, sy = fy * g, sz = fz * g;
            const float bxf = floorf(sx), byf = floorf(sy), bzf = floorf(sz);
            const float tx = sx - bxf, ty = sy - byf, tz = sz - bzf;
            const unsigned bx = (unsigned)bxf;
            const unsigned by = (unsigned)byf;
            const unsigned bz = (unsigned)bzf;
            // prime-multiplied corner hashes (uint32 wrap), incremental for +1
            const unsigned hx0 = bx;                   const unsigned hx1 = bx + 1u;
            const unsigned hy0 = by * 2654435761u;     const unsigned hy1 = hy0 + 2654435761u;
            const unsigned hz0 = bz * 805459861u;      const unsigned hz1 = hz0 + 805459861u;

            const float2* tl = reinterpret_cast<const float2*>(table) + (size_t)l * TSIZE;

            float f0 = 0.f, f1 = 0.f;
            #pragma unroll
            for (int c = 0; c < 8; ++c) {
                const unsigned hx = (c & 4) ? hx1 : hx0;
                const unsigned hy = (c & 2) ? hy1 : hy0;
                const unsigned hz = (c & 1) ? hz1 : hz0;
                const unsigned idx = (hx ^ hy ^ hz) & (TSIZE - 1);
                const float2 v = tl[idx];
                const float w = ((c & 4) ? tx : 1.f - tx) *
                                ((c & 2) ? ty : 1.f - ty) *
                                ((c & 1) ? tz : 1.f - tz);
                f0 = fmaf(w, v.x, f0);
                f1 = fmaf(w, v.y, f1);
            }
            // pack the level's two f16 features into one dword store
            union { _Float16 h2[2]; unsigned u; } pk;
            pk.h2[0] = (_Float16)f0;
            pk.h2[1] = (_Float16)f1;
            sA32[t * (KPAD / 2) + l] = pk.u;
        }
        #pragma unroll
        for (int q = DIN / 2; q < KPAD / 2; ++q) sA32[t * (KPAD / 2) + q] = 0u;
    } else {
        #pragma unroll
        for (int q = 0; q < KPAD / 2; ++q) sA32[t * (KPAD / 2) + q] = 0u;
    }

    __syncthreads();

    // ---------------- Phase 2: fused MLP via WMMA ----------------
    const int wave = t >> 5;
    const int lane = t & 31;
    const int n16  = lane & 15;   // column (B/C) / row (A) selector within tile
    const int h    = lane >> 4;   // half-wave selector

    // B fragments from LDS-staged W1 (f32 -> f16, K padded to 32).
    // 16-bit B 32x16 layout mirrors A: lane half h holds K in {8h..8h+7} u {16+8h..23+8h}.
    v16h Bm[4];
    #pragma unroll
    for (int nt = 0; nt < 4; ++nt) {
        const int n = nt * 16 + n16;
        #pragma unroll
        for (int v = 0; v < 8; ++v) {
            const int kbase = ((v < 4) ? (2 * v) : (16 + 2 * (v - 4))) + 8 * h;
            #pragma unroll
            for (int e = 0; e < 2; ++e) {
                const int k = kbase + e;
                Bm[nt][2 * v + e] = (k < DIN) ? (_Float16)sW1[k * WIDTH + n]
                                              : (_Float16)0.f;
            }
        }
    }

    const float b2v = b2[0];
    float biasv[4], w2v[4];
    #pragma unroll
    for (int nt = 0; nt < 4; ++nt) {
        biasv[nt] = sB1W2[nt * 16 + n16];
        w2v[nt]   = sB1W2[WIDTH + nt * 16 + n16];
    }

    const int b0 = n16 & 1;
    const int b1b = (n16 >> 1) & 1;
    const int b2b = (n16 >> 2) & 1;
    const int rfin = 4 * b0 + 2 * b1b + b2b;   // output row this lane ends up owning

    // Each wave handles two 16-point M-tiles of this block's 256 points.
    #pragma unroll
    for (int mt = 0; mt < 2; ++mt) {
        const int tile = wave * 2 + mt;
        const int row  = tile * 16 + n16;   // point row inside block (A: M = n16)

        // A fragment: two 16B LDS loads per lane per the 16-bit A 16x32 layout
        const v8h lo = *reinterpret_cast<const v8h*>(&sA[row * KPAD + 8 * h]);
        const v8h hi = *reinterpret_cast<const v8h*>(&sA[row * KPAD + 16 + 8 * h]);
        v16h A;
        #pragma unroll
        for (int e = 0; e < 8; ++e) { A[e] = lo[e]; A[8 + e] = hi[e]; }

        float acc[8];
        #pragma unroll
        for (int r = 0; r < 8; ++r) acc[r] = 0.f;

        #pragma unroll
        for (int nt = 0; nt < 4; ++nt) {
            v8f Cz = {};
            // D = A(16x32) x B(32x16) + 0   -> hidden activations tile
            v8f C = __builtin_amdgcn_wmma_f32_16x16x32_f16(
                        /*neg_a=*/false, A, /*neg_b=*/false, Bm[nt],
                        /*c_mod=*/(short)0, Cz, /*reuse_a=*/false, /*reuse_b=*/false);
            // Fused layer 2: relu(C + b1) dot W2, partial per lane.
            // C/D layout: VGPR r, half h -> M = r + 8h ; N = n16 + 16*nt
            #pragma unroll
            for (int r = 0; r < 8; ++r) {
                float hv = C[r] + biasv[nt];
                hv = fmaxf(hv, 0.f);
                acc[r] = fmaf(hv, w2v[nt], acc[r]);
            }
        }

        // Register-halving butterfly over the 16 N-lanes of each half-wave:
        // 8 ds_swizzle total (vs 32 bpermute). xor masks 1/2/4/8 stay in-half.
        float t4[4];
        #pragma unroll
        for (int j = 0; j < 4; ++j) {
            const float send = b0 ? acc[j]     : acc[j + 4];
            const float keep = b0 ? acc[j + 4] : acc[j];
            t4[j] = keep + SWZ_XOR(send, 1);
        }
        float u2[2];
        #pragma unroll
        for (int j = 0; j < 2; ++j) {
            const float send = b1b ? t4[j]     : t4[j + 2];
            const float keep = b1b ? t4[j + 2] : t4[j];
            u2[j] = keep + SWZ_XOR(send, 2);
        }
        {
            const float send = b2b ? u2[0] : u2[1];
            const float keep = b2b ? u2[1] : u2[0];
            float vfin = keep + SWZ_XOR(send, 4);
            vfin += SWZ_XOR(vfin, 8);      // lanes differing in bit3 hold same r
            if (!(n16 & 8)) {              // one writer per (r, half)
                const int pp = pbase + tile * 16 + rfin + 8 * h;
                if (pp < N) out[pp] = vfin + b2v;
            }
        }
    }
}

extern "C" void kernel_launch(void* const* d_in, const int* in_sizes, int n_in,
                              void* d_out, int out_size, void* d_ws, size_t ws_size,
                              hipStream_t stream)
{
    const float* pos   = (const float*)d_in[0];  // [N,3]
    const float* table = (const float*)d_in[1];  // [10,65536,2]
    const float* W1    = (const float*)d_in[2];  // [20,64]
    const float* b1    = (const float*)d_in[3];  // [64]
    const float* W2    = (const float*)d_in[4];  // [64,1]
    const float* b2    = (const float*)d_in[5];  // [1]
    float* out = (float*)d_out;                  // [N,1]

    const int N = in_sizes[0] / 3;
    const int blocks = (N + BPTS - 1) / BPTS;
    ngp_fused_kernel<<<blocks, BPTS, 0, stream>>>(pos, table, W1, b1, W2, b2, out, N);
}
